// TextEncodeBlock_42623255446296
// MI455X (gfx1250) — compile-verified, matrix operand order checked by reading, not run on hardware
//
#include <hip/hip_runtime.h>
#include <hip/hip_bf16.h>

// ---------------------------------------------------------------------------
// MI455X (gfx1250) implementation: bf16 WMMA GEMMs + persistent LSTM
// recurrence with LDS-resident recurrent weights and grid-wide barriers.
// ---------------------------------------------------------------------------

typedef __attribute__((ext_vector_type(16))) __bf16 v16bf;
typedef __attribute__((ext_vector_type(8)))  __bf16 v8bf;
typedef __attribute__((ext_vector_type(8)))  float  v8f;

static constexpr int cB = 128;
static constexpr int cT = 256;
static constexpr int cF = 768;
static constexpr int cH = 512;

// ---------------------------------------------------------------------------
// helpers
// ---------------------------------------------------------------------------
__device__ __forceinline__ v16bf make_v16(v8bf lo, v8bf hi) {
  v16bf a;
#pragma unroll
  for (int i = 0; i < 8; ++i) { a[i] = lo[i]; a[i + 8] = hi[i]; }
  return a;
}

__device__ __forceinline__ float sigmoidf_fast(float x) {
  return 1.0f / (1.0f + __expf(-x));
}

// sense-reversing grid barrier (nblk block leaders arrive; others spin on gen)
__device__ __forceinline__ void grid_sync(unsigned* cnt, unsigned* gen, unsigned nblk) {
  __threadfence();
  __syncthreads();
  if (threadIdx.x == 0) {
    volatile unsigned* vgen = gen;
    unsigned g = *vgen;
    if (atomicAdd(cnt, 1u) == nblk - 1u) {
      atomicExch(cnt, 0u);
      __threadfence();
      atomicAdd(gen, 1u);
    } else {
      while (*vgen == g) { __builtin_amdgcn_s_sleep(2); }
    }
  }
  __syncthreads();
  __threadfence();
}

// ---------------------------------------------------------------------------
// fp32 -> bf16 convert / transpose-convert (weight staging, runs once per call)
// ---------------------------------------------------------------------------
__global__ void cvt_f32_bf16(const float* __restrict__ src, __bf16* __restrict__ dst, size_t n) {
  size_t i = (size_t)blockIdx.x * blockDim.x + threadIdx.x;
  if (i < n) dst[i] = (__bf16)src[i];
}

__global__ void transpose_cvt(const float* __restrict__ src, __bf16* __restrict__ dst,
                              int R, int C) {
  size_t i = (size_t)blockIdx.x * blockDim.x + threadIdx.x;
  if (i < (size_t)R * C) {
    int r = (int)(i / C);
    int c = (int)(i % C);
    dst[(size_t)c * R + r] = (__bf16)src[i];
  }
}

__global__ void zero_state(__bf16* __restrict__ hcur, unsigned* __restrict__ bar, int n) {
  int i = blockIdx.x * blockDim.x + threadIdx.x;
  if (i < n) hcur[i] = (__bf16)0.0f;
  if (i < 2) bar[i] = 0u;
}

// ---------------------------------------------------------------------------
// Generic bf16 WMMA GEMM:  C[M,N] = act(A[M,K] @ B^T[N,K] + bias)
// B is stored N-major, K-contiguous (each lane loads 16B runs of K).
// Block = 8 waves; each wave computes a 16(M) x 64(N) strip.
// grid = (N/64, M/128)
// ---------------------------------------------------------------------------
template <int ACT, bool OUT_BF16>
__global__ __launch_bounds__(256) void gemm_wmma(
    const __bf16* __restrict__ A, const __bf16* __restrict__ Bm,
    const float* __restrict__ bias, void* __restrict__ Cout,
    int M, int N, int K) {
  const int lane = threadIdx.x & 31;
  const int wave = threadIdx.x >> 5;
  const int q = lane & 15;        // A: row M ; B: col N ; C/D: col N
  const int half = lane >> 4;     // K-half selector for A/B, M-half for C/D
  const int m0 = (blockIdx.y * 8 + wave) * 16;
  const int n0 = blockIdx.x * 64;

  v8f acc[4] = {};

  const __bf16* arow = A + (size_t)(m0 + q) * K;
  for (int kb = 0; kb < K; kb += 32) {
    const int kb0 = kb + half * 8;
    v16bf a = make_v16(*(const v8bf*)(arow + kb0), *(const v8bf*)(arow + kb0 + 16));
#pragma unroll
    for (int j = 0; j < 4; ++j) {
      const __bf16* brow = Bm + (size_t)(n0 + j * 16 + q) * K + kb0;
      v16bf b = make_v16(*(const v8bf*)brow, *(const v8bf*)(brow + 16));
      acc[j] = __builtin_amdgcn_wmma_f32_16x16x32_bf16(
          false, a, false, b, (short)0, acc[j], false, false);
    }
  }

#pragma unroll
  for (int j = 0; j < 4; ++j) {
    const int n = n0 + j * 16 + q;
    const float bv = bias ? bias[n] : 0.0f;
#pragma unroll
    for (int r = 0; r < 8; ++r) {
      const int m = m0 + r + 8 * half;
      float v = acc[j][r] + bv;
      if (ACT == 1) v = (v > 0.0f) ? v : 0.01f * v;  // LeakyReLU(0.01)
      const size_t idx = (size_t)m * N + n;
      if (OUT_BF16) ((__bf16*)Cout)[idx] = (__bf16)v;
      else          ((float*)Cout)[idx]  = v;
    }
  }
}

// ---------------------------------------------------------------------------
// Persistent LSTM recurrence.
// grid = 32 blocks (one per 16 hidden cols), 256 threads (8 waves = all 128 B rows).
// Whh (4 gates x 16 cols x 512 K) converted to bf16 and kept in LDS (64 KB,
// XOR chunk-swizzled so 16-byte ds_load_b128 per lane is bank-conflict free).
// Cell state c lives in registers (WMMA C/D layout) across all T steps.
// h ping-pongs through a small global bf16 buffer; one grid barrier per step.
// ---------------------------------------------------------------------------
__device__ __forceinline__ int swz(int j, int k) {
  // element offset within one [16][512] bf16 plane; swizzle 16B chunks
  const int chunk = k >> 3;             // 0..63
  const int cc = chunk ^ j;             // j in 0..15
  return j * 512 + cc * 8 + (k & 7);
}

template <bool STORE_SEQ>
__global__ __launch_bounds__(256) void lstm_recur(
    const float* __restrict__ xg,       // [B, T, 4H] fp32 (includes bih)
    const float* __restrict__ Whh,      // [4H, H] fp32
    const float* __restrict__ bhh,      // [4H]
    const int* __restrict__ tlen,       // [B] (used when !STORE_SEQ)
    __bf16* __restrict__ hseq,          // [B*T, H] bf16 (when STORE_SEQ)
    __bf16* __restrict__ hcur,          // [2, B, H] bf16 ping-pong, zeroed
    __bf16* __restrict__ hlast,         // [B, H] bf16 (when !STORE_SEQ)
    unsigned* __restrict__ bar,         // {cnt, gen} zeroed
    int T) {
  extern __shared__ __bf16 sW[];        // 4 * 16 * 512 bf16 = 64 KB
  const unsigned nwg = gridDim.x;       // 32
  const int n0 = blockIdx.x * 16;       // hidden-column base
  const int wave = threadIdx.x >> 5;
  const int lane = threadIdx.x & 31;
  const int q = lane & 15;
  const int half = lane >> 4;
  const int mb = wave * 16;             // batch-row base for this wave

  // --- preload recurrent weights to LDS (f32 -> bf16, swizzled) ---
  for (int i = threadIdx.x; i < 4 * 16 * cH; i += 256) {
    const int g = i >> 13;              // / 8192
    const int rem = i & 8191;
    const int j = rem >> 9;             // / 512
    const int k = rem & 511;
    sW[g * (16 * 512) + swz(j, k)] =
        (__bf16)Whh[(size_t)(g * cH + n0 + j) * cH + k];
  }
  __syncthreads();

  float bh[4];
#pragma unroll
  for (int g = 0; g < 4; ++g) bh[g] = bhh[g * cH + n0 + q];

  int tl[8];
  if (!STORE_SEQ) {
#pragma unroll
    for (int r = 0; r < 8; ++r) tl[r] = tlen[mb + r + 8 * half] - 1;
  }

  float cstate[8];
#pragma unroll
  for (int r = 0; r < 8; ++r) cstate[r] = 0.0f;

  for (int t = 0; t < T; ++t) {
    const __bf16* hin = hcur + (size_t)(t & 1) * cB * cH;
    __bf16* hout = hcur + (size_t)((t + 1) & 1) * cB * cH;

    v8f acc[4] = {};
    const __bf16* arow = hin + (size_t)(mb + q) * cH;
#pragma unroll 4
    for (int kb = 0; kb < cH; kb += 32) {
      const int kb0 = kb + half * 8;
      v16bf a = make_v16(*(const v8bf*)(arow + kb0), *(const v8bf*)(arow + kb0 + 16));
#pragma unroll
      for (int g = 0; g < 4; ++g) {
        const __bf16* bp0 = sW + g * (16 * 512) + swz(q, kb0);
        const __bf16* bp1 = sW + g * (16 * 512) + swz(q, kb0 + 16);
        v16bf b = make_v16(*(const v8bf*)bp0, *(const v8bf*)bp1);
        acc[g] = __builtin_amdgcn_wmma_f32_16x16x32_bf16(
            false, a, false, b, (short)0, acc[g], false, false);
      }
    }

    // gates: i, f, g, o  (PyTorch order) ; xg already holds x@Wih^T + bih
#pragma unroll
    for (int r = 0; r < 8; ++r) {
      const int b = mb + r + 8 * half;
      const float* xp = xg + ((size_t)b * T + t) * (4 * cH) + n0 + q;
      const float gi = acc[0][r] + xp[0]        + bh[0];
      const float gf = acc[1][r] + xp[cH]       + bh[1];
      const float gg = acc[2][r] + xp[2 * cH]   + bh[2];
      const float go = acc[3][r] + xp[3 * cH]   + bh[3];
      const float c = sigmoidf_fast(gf) * cstate[r] + sigmoidf_fast(gi) * tanhf(gg);
      cstate[r] = c;
      const float h = sigmoidf_fast(go) * tanhf(c);
      const __bf16 hb = (__bf16)h;
      hout[(size_t)b * cH + n0 + q] = hb;
      if (STORE_SEQ) {
        hseq[((size_t)b * T + t) * cH + n0 + q] = hb;
      } else if (t == tl[r]) {
        hlast[(size_t)b * cH + n0 + q] = hb;
      }
    }
    grid_sync(bar, bar + 1, nwg);
  }
}

// ---------------------------------------------------------------------------
// host-side orchestration
// ---------------------------------------------------------------------------
extern "C" void kernel_launch(void* const* d_in, const int* in_sizes, int n_in,
                              void* d_out, int out_size, void* d_ws, size_t ws_size,
                              hipStream_t stream) {
  (void)in_sizes; (void)n_in; (void)out_size; (void)ws_size;
  const float* emb  = (const float*)d_in[0];
  const int*   tlen = (const int*)  d_in[1];
  const float* W1   = (const float*)d_in[2];
  const float* b1   = (const float*)d_in[3];
  const float* W2   = (const float*)d_in[4];
  const float* b2   = (const float*)d_in[5];
  const float* Wih0 = (const float*)d_in[6];
  const float* Whh0 = (const float*)d_in[7];
  const float* bih0 = (const float*)d_in[8];
  const float* bhh0 = (const float*)d_in[9];
  const float* Wih1 = (const float*)d_in[10];
  const float* Whh1 = (const float*)d_in[11];
  const float* bih1 = (const float*)d_in[12];
  const float* bhh1 = (const float*)d_in[13];
  const float* Wl   = (const float*)d_in[14];
  const float* bl   = (const float*)d_in[15];
  float* out = (float*)d_out;

  char* ws = (char*)d_ws;
  size_t off = 0;
  auto alloc = [&](size_t bytes) -> void* {
    void* p = ws + off;
    off = (off + bytes + 255) & ~(size_t)255;
    return p;
  };

  const size_t nEmb = (size_t)cB * cT * cF;       // 25,165,824
  const size_t nBT  = (size_t)cB * cT;            // 32,768 rows
  __bf16* embB  = (__bf16*)alloc(nEmb * 2);
  __bf16* w1t   = (__bf16*)alloc((size_t)cF * cH * 2);
  __bf16* w2t   = (__bf16*)alloc((size_t)cH * cH * 2);
  __bf16* wih0b = (__bf16*)alloc((size_t)4 * cH * cH * 2);
  __bf16* wih1b = (__bf16*)alloc((size_t)4 * cH * cH * 2);
  __bf16* wlb   = (__bf16*)alloc((size_t)cH * cH * 2);
  __bf16* tmp1  = (__bf16*)alloc(nBT * cH * 2);
  __bf16* xbuf  = (__bf16*)alloc(nBT * cH * 2);
  __bf16* hseq  = (__bf16*)alloc(nBT * cH * 2);
  float*  xg    = (float*) alloc(nBT * (size_t)(4 * cH) * 4);   // 256 MB, reused
  __bf16* hcur  = (__bf16*)alloc((size_t)2 * cB * cH * 2);
  __bf16* hlast = (__bf16*)alloc((size_t)cB * cH * 2);
  unsigned* bar = (unsigned*)alloc(256);

  const int TPB = 256;
  auto blk = [](size_t n, int t) { return (unsigned)((n + t - 1) / t); };

  // stage weights / activations into bf16 WMMA layouts
  cvt_f32_bf16<<<blk(nEmb, TPB), TPB, 0, stream>>>(emb, embB, nEmb);
  transpose_cvt<<<blk((size_t)cF * cH, TPB), TPB, 0, stream>>>(W1, w1t, cF, cH);
  transpose_cvt<<<blk((size_t)cH * cH, TPB), TPB, 0, stream>>>(W2, w2t, cH, cH);
  cvt_f32_bf16<<<blk((size_t)4 * cH * cH, TPB), TPB, 0, stream>>>(Wih0, wih0b, (size_t)4 * cH * cH);
  cvt_f32_bf16<<<blk((size_t)4 * cH * cH, TPB), TPB, 0, stream>>>(Wih1, wih1b, (size_t)4 * cH * cH);
  cvt_f32_bf16<<<blk((size_t)cH * cH, TPB), TPB, 0, stream>>>(Wl, wlb, (size_t)cH * cH);

  // MLP: tmp1 = leaky_relu(emb @ W1 + b1);  x = tmp1 @ W2 + b2
  gemm_wmma<1, true><<<dim3(cH / 64, (unsigned)(nBT / 128)), TPB, 0, stream>>>(
      embB, w1t, b1, tmp1, (int)nBT, cH, cF);
  gemm_wmma<0, true><<<dim3(cH / 64, (unsigned)(nBT / 128)), TPB, 0, stream>>>(
      tmp1, w2t, b2, xbuf, (int)nBT, cH, cH);

  // layer 0: xg = x @ Wih0^T + bih0 ; recurrence -> hseq
  gemm_wmma<0, false><<<dim3(4 * cH / 64, (unsigned)(nBT / 128)), TPB, 0, stream>>>(
      xbuf, wih0b, bih0, xg, (int)nBT, 4 * cH, cH);
  zero_state<<<blk((size_t)2 * cB * cH, TPB), TPB, 0, stream>>>(hcur, bar, 2 * cB * cH);
  lstm_recur<true><<<dim3(cH / 16), TPB, 4 * 16 * cH * sizeof(__bf16), stream>>>(
      xg, Whh0, bhh0, nullptr, hseq, hcur, nullptr, bar, cT);

  // layer 1: xg = hseq @ Wih1^T + bih1 ; recurrence -> hlast (at len-1)
  gemm_wmma<0, false><<<dim3(4 * cH / 64, (unsigned)(nBT / 128)), TPB, 0, stream>>>(
      hseq, wih1b, bih1, xg, (int)nBT, 4 * cH, cH);
  zero_state<<<blk((size_t)2 * cB * cH, TPB), TPB, 0, stream>>>(hcur, bar, 2 * cB * cH);
  lstm_recur<false><<<dim3(cH / 16), TPB, 4 * 16 * cH * sizeof(__bf16), stream>>>(
      xg, Whh1, bhh1, tlen, nullptr, hcur, hlast, bar, cT);

  // out = hlast @ Wl^T + bl   (M=128 -> single row-block)
  gemm_wmma<0, false><<<dim3(cH / 64, cB / 128), TPB, 0, stream>>>(
      hlast, wlb, bl, out, cB, cH, cH);
}
